// ContinuousMambaBlock_21904333209606
// MI455X (gfx1250) — compile-verified, matrix operand order checked by reading, not run on hardware
//
#include <hip/hip_runtime.h>
#include <math.h>

#define B_       2
#define L_       2048
#define D_MODEL  1024
#define D_INNER  2048
#define D_STATE  128
#define D_CONV   4
#define NHEADS   32
#define HEADDIM  64
#define D_TIME   64
#define CONV_DIM (D_INNER + 2 * D_STATE)               /* 2304 */
#define D_IN_PROJ (2 * D_INNER + 2 * D_STATE + NHEADS) /* 4384 */
#define BL       (B_ * L_)                             /* 4096 */
#define EPS_     1e-5f

typedef __attribute__((ext_vector_type(16))) __bf16 v16bf;
typedef __attribute__((ext_vector_type(8)))  __bf16 v8bf;
typedef __attribute__((ext_vector_type(8)))  float  v8f;
typedef __attribute__((ext_vector_type(4)))  float  v4f;
typedef __attribute__((ext_vector_type(4)))  int    v4i;

#if defined(__has_builtin)
#  if __has_builtin(__builtin_amdgcn_global_load_async_to_lds_b128)
#    define HAVE_ASYNC_LDS 1
#  endif
#endif

__device__ __forceinline__ unsigned short f2bf_bits(float f) {
    unsigned u = __builtin_bit_cast(unsigned, f);
    u += 0x7FFFu + ((u >> 16) & 1u);        // round-to-nearest-even
    return (unsigned short)(u >> 16);
}

__device__ __forceinline__ v16bf cat16(v8bf lo, v8bf hi) {
    return __builtin_shufflevector(lo, hi, 0,1,2,3,4,5,6,7,8,9,10,11,12,13,14,15);
}

__device__ __forceinline__ void copy16B_to_lds(const unsigned short* gp,
                                               unsigned short* lp) {
#ifdef HAVE_ASYNC_LDS
    __builtin_amdgcn_global_load_async_to_lds_b128(
        (__attribute__((address_space(1))) v4i*)gp,
        (__attribute__((address_space(3))) v4i*)lp, 0, 0);
#else
    *(v8bf*)lp = *(const v8bf*)gp;
#endif
}

__device__ __forceinline__ void wait_async_lds() {
#ifdef HAVE_ASYNC_LDS
#  if __has_builtin(__builtin_amdgcn_s_wait_asynccnt)
    __builtin_amdgcn_s_wait_asynccnt(0);
#  else
    asm volatile("s_wait_asynccnt 0" ::: "memory");
#  endif
#endif
}

// ---------------------------------------------------------------------------
// one-shot f32 -> bf16 conversion (A-side activations), vectorized 4/thread
// ---------------------------------------------------------------------------
__global__ __launch_bounds__(256)
void f32_to_bf16_kernel(const float* __restrict__ in,
                        unsigned short* __restrict__ out, long n)
{
    long i = ((long)blockIdx.x * 256 + threadIdx.x) * 4;
    if (i + 3 < n) {
        v4f v = *(const v4f*)(in + i);
        unsigned r0 = (unsigned)f2bf_bits(v[0]) | ((unsigned)f2bf_bits(v[1]) << 16);
        unsigned r1 = (unsigned)f2bf_bits(v[2]) | ((unsigned)f2bf_bits(v[3]) << 16);
        unsigned* o = (unsigned*)(out + i);
        o[0] = r0; o[1] = r1;
    } else {
        for (; i < n; ++i) out[i] = f2bf_bits(in[i]);
    }
}

// ---------------------------------------------------------------------------
// one-shot weight prep: f32 [K][N] -> bf16 transposed [N][K], LDS-tiled 32x32
// so both global sides are coalesced.  Done once; makes GEMM B-staging a pure
// contiguous 16B async copy (same as A).
// ---------------------------------------------------------------------------
__global__ __launch_bounds__(256)
void f32_to_bf16t_kernel(const float* __restrict__ in,
                         unsigned short* __restrict__ out, int K, int N)
{
    __shared__ float tile[32][33];
    const int k0 = blockIdx.x * 32;
    const int n0 = blockIdx.y * 32;
    const int tx = threadIdx.x & 31;
    const int ty = threadIdx.x >> 5;           // 0..7
    #pragma unroll
    for (int i = ty; i < 32; i += 8) {
        int k = k0 + i, n = n0 + tx;
        tile[i][tx] = (k < K && n < N) ? in[(long)k * N + n] : 0.0f;
    }
    __syncthreads();
    #pragma unroll
    for (int i = ty; i < 32; i += 8) {
        int n = n0 + i, k = k0 + tx;
        if (n < N && k < K) out[(long)n * K + k] = f2bf_bits(tile[tx][i]);
    }
}

// ---------------------------------------------------------------------------
// bf16 WMMA GEMM:  C[M,N] = A[M,K](bf16 row-major) @ Bt[N,K](bf16, B transposed)
// WG = 256 threads (8 waves). WG tile 128x128, wave tile 32x64 (2x4 frags)
//  -> 8 v_wmma_f32_16x16x32_bf16 per K-step of 32.
// Requirements (guaranteed by launches): M % 128 == 0, K % 32 == 0.
// Both tiles staged with per-lane GLOBAL_LOAD_ASYNC_TO_LDS_B128 (ASYNCcnt);
// out-of-range B rows are clamped (their outputs are discarded at store).
// Fragment gather per ISA 16-bit layout: lane g=lane/16 takes K in
// [8g..8g+7] and [16+8g..16+8g+7] -> two contiguous ds_load_b128.
// ---------------------------------------------------------------------------
#define TM 128
#define TN 128
#define TK 32

template <bool OUTBF>
__global__ __launch_bounds__(256)
void gemm_bf16_wmma(const unsigned short* __restrict__ A,
                    const unsigned short* __restrict__ Bt,
                    const float* __restrict__ bias,
                    void* __restrict__ Cptr,
                    int M, int N, int K)
{
    __shared__ __align__(16) unsigned short sA[TM * TK]; // 8 KB  [m][k]
    __shared__ __align__(16) unsigned short sB[TN * TK]; // 8 KB  [n][k]

    const int tid  = threadIdx.x;
    const int wave = tid >> 5;
    const int lane = tid & 31;
    const int wm   = (wave >> 1) * 32;   // 4 wave-rows
    const int wn   = (wave & 1) * 64;    // 2 wave-cols
    const int bm   = blockIdx.x * TM;
    const int bn   = blockIdx.y * TN;
    const int g    = lane >> 4;
    const int r    = lane & 15;

    v8f acc[2][4] = {};

    for (int k0 = 0; k0 < K; k0 += TK) {
        // A tile: 128 rows x 64B = 512 x 16B chunks, 2 per thread
        #pragma unroll
        for (int c = tid; c < 512; c += 256) {
            int m  = c >> 2;
            int qd = (c & 3) * 8;
            copy16B_to_lds(A + (long)(bm + m) * K + k0 + qd, &sA[m * 32 + qd]);
        }
        // B tile (already transposed in global): identical addressing
        #pragma unroll
        for (int c = tid; c < 512; c += 256) {
            int n  = c >> 2;
            int qd = (c & 3) * 8;
            int gn = bn + n;
            gn = (gn < N) ? gn : 0;     // clamp; clamped cols never stored
            copy16B_to_lds(Bt + (long)gn * K + k0 + qd, &sB[n * 32 + qd]);
        }
        wait_async_lds();
        __syncthreads();

        v16bf af[2], bfr[4];
        #pragma unroll
        for (int i = 0; i < 2; ++i) {
            int m = wm + i * 16 + r;
            af[i] = cat16(*(const v8bf*)&sA[m * 32 + g * 8],
                          *(const v8bf*)&sA[m * 32 + 16 + g * 8]);
        }
        #pragma unroll
        for (int j = 0; j < 4; ++j) {
            int n = wn + j * 16 + r;
            bfr[j] = cat16(*(const v8bf*)&sB[n * 32 + g * 8],
                           *(const v8bf*)&sB[n * 32 + 16 + g * 8]);
        }

        #pragma unroll
        for (int i = 0; i < 2; ++i)
            #pragma unroll
            for (int j = 0; j < 4; ++j)
                acc[i][j] = __builtin_amdgcn_wmma_f32_16x16x32_bf16(
                    false, af[i], false, bfr[j], (short)0, acc[i][j],
                    false, false);
        __syncthreads();
    }

    // C/D layout: lane, VGPR v -> M = v + 8*(lane/16), N = lane%16
    #pragma unroll
    for (int i = 0; i < 2; ++i) {
        #pragma unroll
        for (int j = 0; j < 4; ++j) {
            int n = bn + wn + j * 16 + r;
            if (n >= N) continue;
            float bv = bias ? bias[n] : 0.0f;
            #pragma unroll
            for (int v = 0; v < 8; ++v) {
                int m = bm + wm + i * 16 + v + 8 * g;
                float val = acc[i][j][v] + bv;
                if (OUTBF) ((unsigned short*)Cptr)[(long)m * N + n] = f2bf_bits(val);
                else       ((float*)Cptr)[(long)m * N + n] = val;
            }
        }
    }
}

// ---------------------------------------------------------------------------
// causal depthwise conv (D_CONV=4) over xBC slice of zxbcdt, then SiLU
// ---------------------------------------------------------------------------
__global__ __launch_bounds__(256)
void conv_silu_kernel(const float* __restrict__ zxbcdt,
                      const float* __restrict__ conv_w,
                      const float* __restrict__ conv_b,
                      float* __restrict__ xbc)
{
    long idx = (long)blockIdx.x * 256 + threadIdx.x;
    if (idx >= (long)BL * CONV_DIM) return;
    int c = (int)(idx % CONV_DIM);
    long t = idx / CONV_DIM;
    int l = (int)(t % L_);
    int b = (int)(t / L_);
    float acc = conv_b[c];
    #pragma unroll
    for (int k = 0; k < D_CONV; ++k) {
        int ls = l - (D_CONV - 1) + k;
        if (ls >= 0)
            acc = fmaf(zxbcdt[((long)b * L_ + ls) * D_IN_PROJ + D_INNER + c],
                       conv_w[k * CONV_DIM + c], acc);
    }
    xbc[idx] = acc / (1.0f + __expf(-acc));   // silu
}

// ---------------------------------------------------------------------------
// dt = softplus(dt_raw + dt_bias + dtmod); dA = exp(dt * (-exp(A_log)))
// ---------------------------------------------------------------------------
__global__ __launch_bounds__(256)
void dt_prep_kernel(const float* __restrict__ zxbcdt,
                    const float* __restrict__ dtmod,
                    const float* __restrict__ dt_bias,
                    const float* __restrict__ A_log,
                    float* __restrict__ dtb, float* __restrict__ dAb)
{
    int idx = blockIdx.x * 256 + threadIdx.x;
    if (idx >= BL * NHEADS) return;
    int hh = idx & (NHEADS - 1);
    long row = idx >> 5;
    float raw = zxbcdt[row * D_IN_PROJ + D_INNER + CONV_DIM + hh]
              + dt_bias[hh] + dtmod[idx];
    float dtv = (raw > 20.0f) ? raw : log1pf(__expf(raw));
    float Ah  = -__expf(A_log[hh]);
    dtb[idx] = dtv;
    dAb[idx] = __expf(dtv * Ah);
}

// ---------------------------------------------------------------------------
// selective scan: one WG per (batch, head). 256 thr: thread=(p, n-quarter),
// 32 fp32 state regs/thread; B_t/C_t double-buffered in LDS (1 barrier/step);
// y reduced over n with two wave32 shfl_xor hops.
// ---------------------------------------------------------------------------
__global__ __launch_bounds__(256)
void scan_kernel(const float* __restrict__ xbc,
                 const float* __restrict__ dtb,
                 const float* __restrict__ dAb,
                 float* __restrict__ ybuf)
{
    const int b  = blockIdx.x >> 5;
    const int hh = blockIdx.x & 31;
    const int tid = threadIdx.x;
    const int p  = tid >> 2;
    const int q  = tid & 3;
    const int n0 = q * 32;

    __shared__ float sB[2][D_STATE];
    __shared__ float sC[2][D_STATE];

    float hstate[32];
    #pragma unroll
    for (int i = 0; i < 32; ++i) hstate[i] = 0.0f;

    for (int l = 0; l < L_; ++l) {
        const long row = (long)b * L_ + l;
        const float* xr = xbc + row * CONV_DIM;
        const int s = l & 1;
        if (tid < D_STATE)          sB[s][tid] = xr[D_INNER + tid];
        else if (tid < 2 * D_STATE) sC[s][tid - D_STATE] = xr[D_INNER + tid];
        const float dt_t = dtb[row * NHEADS + hh];
        const float dA_t = dAb[row * NHEADS + hh];
        const float x_t  = xr[hh * HEADDIM + p];
        __syncthreads();

        const float coef = dt_t * x_t;
        float y = 0.0f;
        const float* Bp = &sB[s][n0];
        const float* Cp = &sC[s][n0];
        #pragma unroll
        for (int i = 0; i < 32; ++i) {
            hstate[i] = fmaf(dA_t, hstate[i], coef * Bp[i]);
            y = fmaf(hstate[i], Cp[i], y);
        }
        y += __shfl_xor(y, 1);
        y += __shfl_xor(y, 2);
        if (q == 0) ybuf[row * D_INNER + hh * HEADDIM + p] = y;
    }
}

// ---------------------------------------------------------------------------
// y = (scan_y + x*D) * silu(z); RMS-norm over D_INNER; emit bf16 for final GEMM
// ---------------------------------------------------------------------------
__global__ __launch_bounds__(256)
void gate_norm_kernel(const float* __restrict__ zxbcdt,
                      const float* __restrict__ xbc,
                      const float* __restrict__ Dv,
                      const float* __restrict__ norm_w,
                      const float* __restrict__ ybuf,
                      unsigned short* __restrict__ ynorm)
{
    const long row = blockIdx.x;
    const float* zr = zxbcdt + row * D_IN_PROJ;
    const float* xr = xbc + row * CONV_DIM;
    const float* yr = ybuf + row * D_INNER;
    unsigned short* yo = ynorm + row * D_INNER;
    const int tid = threadIdx.x;

    float vals[8];
    float ss = 0.0f;
    #pragma unroll
    for (int j = 0; j < 8; ++j) {
        int i = tid + j * 256;
        int hh = i >> 6;                            // / HEADDIM
        float yv = yr[i] + xr[i] * Dv[hh];
        float z  = zr[i];
        float gv = yv * (z / (1.0f + __expf(-z)));
        vals[j] = gv;
        ss += gv * gv;
    }
    __shared__ float red[256];
    red[tid] = ss;
    __syncthreads();
    for (int off = 128; off > 0; off >>= 1) {
        if (tid < off) red[tid] += red[tid + off];
        __syncthreads();
    }
    float scale = rsqrtf(red[0] * (1.0f / D_INNER) + EPS_);
    #pragma unroll
    for (int j = 0; j < 8; ++j) {
        int i = tid + j * 256;
        yo[i] = f2bf_bits(vals[j] * scale * norm_w[i]);
    }
}

// ---------------------------------------------------------------------------
extern "C" void kernel_launch(void* const* d_in, const int* in_sizes, int n_in,
                              void* d_out, int out_size, void* d_ws, size_t ws_size,
                              hipStream_t stream)
{
    const float* hs     = (const float*)d_in[0];
    const float* dte    = (const float*)d_in[1];
    const float* Wproj  = (const float*)d_in[2];
    const float* bproj  = (const float*)d_in[3];
    const float* Wdtmod = (const float*)d_in[4];
    const float* bdtmod = (const float*)d_in[5];
    const float* Win    = (const float*)d_in[6];
    const float* convw  = (const float*)d_in[7];
    const float* convb  = (const float*)d_in[8];
    const float* dtbias = (const float*)d_in[9];
    const float* Alog   = (const float*)d_in[10];
    const float* Dv     = (const float*)d_in[11];
    const float* normw  = (const float*)d_in[12];
    const float* Wout   = (const float*)d_in[13];
    float* out = (float*)d_out;
    (void)ws_size; (void)in_sizes; (void)n_in; (void)out_size;

    char* wsb = (char*)d_ws;
    size_t off = 0;
    auto alloc = [&](size_t bytes) -> char* {
        char* p = wsb + off;
        off += (bytes + 255) & ~(size_t)255;
        return p;
    };
    unsigned short* hs_bf     = (unsigned short*)alloc((size_t)BL * D_MODEL * 2);
    unsigned short* dte_bf    = (unsigned short*)alloc((size_t)BL * D_TIME * 2);
    unsigned short* WprojT    = (unsigned short*)alloc((size_t)D_MODEL * D_MODEL * 2);
    unsigned short* WdtmodT   = (unsigned short*)alloc((size_t)D_TIME * NHEADS * 2);
    unsigned short* WinT      = (unsigned short*)alloc((size_t)D_MODEL * D_IN_PROJ * 2);
    unsigned short* WoutT     = (unsigned short*)alloc((size_t)D_INNER * D_MODEL * 2);
    unsigned short* u_bf      = (unsigned short*)alloc((size_t)BL * D_MODEL * 2);
    float* zxbcdt             = (float*)alloc((size_t)BL * D_IN_PROJ * 4);
    float* dtmod              = (float*)alloc((size_t)BL * NHEADS * 4);
    float* xbc                = (float*)alloc((size_t)BL * CONV_DIM * 4);
    float* dtb                = (float*)alloc((size_t)BL * NHEADS * 4);
    float* dAb                = (float*)alloc((size_t)BL * NHEADS * 4);
    float* ybuf               = (float*)alloc((size_t)BL * D_INNER * 4);
    unsigned short* ynorm_bf  = (unsigned short*)alloc((size_t)BL * D_INNER * 2);

    dim3 blk(256);
    auto cgrid = [](long n) { return dim3((unsigned)((n / 4 + 255) / 256)); };
    auto tgrid = [](int K, int N) {
        return dim3((unsigned)((K + 31) / 32), (unsigned)((N + 31) / 32));
    };

    // one-shot operand prep: A-side plain bf16, B-side bf16 + transpose
    f32_to_bf16_kernel<<<cgrid((long)BL * D_MODEL), blk, 0, stream>>>(hs, hs_bf, (long)BL * D_MODEL);
    f32_to_bf16_kernel<<<cgrid((long)BL * D_TIME), blk, 0, stream>>>(dte, dte_bf, (long)BL * D_TIME);
    f32_to_bf16t_kernel<<<tgrid(D_MODEL, D_MODEL), blk, 0, stream>>>(Wproj, WprojT, D_MODEL, D_MODEL);
    f32_to_bf16t_kernel<<<tgrid(D_TIME, NHEADS), blk, 0, stream>>>(Wdtmod, WdtmodT, D_TIME, NHEADS);
    f32_to_bf16t_kernel<<<tgrid(D_MODEL, D_IN_PROJ), blk, 0, stream>>>(Win, WinT, D_MODEL, D_IN_PROJ);
    f32_to_bf16t_kernel<<<tgrid(D_INNER, D_MODEL), blk, 0, stream>>>(Wout, WoutT, D_INNER, D_MODEL);

    // u = hs @ W_proj + b_proj          (4096 x 1024 x 1024) -> bf16
    gemm_bf16_wmma<true><<<dim3(BL / TM, D_MODEL / TN), blk, 0, stream>>>(
        hs_bf, WprojT, bproj, u_bf, BL, D_MODEL, D_MODEL);

    // dtmod = dte @ W_dtmod + b_dtmod   (4096 x 64 -> 32) -> f32
    gemm_bf16_wmma<false><<<dim3(BL / TM, 1), blk, 0, stream>>>(
        dte_bf, WdtmodT, bdtmod, dtmod, BL, NHEADS, D_TIME);

    // zxbcdt = u @ W_in                 (4096 x 1024 x 4384) -> f32
    gemm_bf16_wmma<false><<<dim3(BL / TM, (D_IN_PROJ + TN - 1) / TN), blk, 0, stream>>>(
        u_bf, WinT, nullptr, zxbcdt, BL, D_IN_PROJ, D_MODEL);

    // causal conv + silu
    conv_silu_kernel<<<((long)BL * CONV_DIM + 255) / 256, blk, 0, stream>>>(
        zxbcdt, convw, convb, xbc);

    // dt / dA precompute
    dt_prep_kernel<<<(BL * NHEADS + 255) / 256, blk, 0, stream>>>(
        zxbcdt, dtmod, dtbias, Alog, dtb, dAb);

    // selective scan (sequential critical path)
    scan_kernel<<<B_ * NHEADS, blk, 0, stream>>>(xbc, dtb, dAb, ybuf);

    // gate + RMS norm -> bf16
    gate_norm_kernel<<<BL, blk, 0, stream>>>(zxbcdt, xbc, Dv, normw, ybuf, ynorm_bf);

    // out = ynorm @ W_out               (4096 x 2048 x 1024) -> f32
    gemm_bf16_wmma<false><<<dim3(BL / TM, D_MODEL / TN), blk, 0, stream>>>(
        ynorm_bf, WoutT, nullptr, out, BL, D_MODEL, D_INNER);
}
